// PatchConv_86397562126524
// MI455X (gfx1250) — matrix operand order repaired and checked by reading, not compiled, not run on hardware
//
#include <hip/hip_runtime.h>

// Flip to 0 if the gfx1250 assembler rejects the async-to-LDS mnemonic.
#define USE_ASYNC_LDS 1

// Problem constants (match reference)
#define Bv   16
#define Cv   64
#define Hv   32
#define Wv   32
#define Ov   128
#define CKKv 576          // C*KH*KW
#define Lv   1024         // oh*ow
#define H2v  32           // HID*2

typedef _Float16 half_t;
typedef __attribute__((ext_vector_type(16))) _Float16 v16h;
typedef __attribute__((ext_vector_type(8)))  _Float16 v8h;
typedef __attribute__((ext_vector_type(8)))  float    v8f;

// ---------------------------------------------------------------------------
// WMMA helper: D(f32 16x16) = A(f16 16x32) * B(f16 32x16) + C
// ---------------------------------------------------------------------------
__device__ __forceinline__ v8f wmma16x16x32(v16h a, v16h b, v8f c) {
  return __builtin_amdgcn_wmma_f32_16x16x32_f16(
      /*neg_a=*/false, a, /*neg_b=*/false, b,
      /*c_mod=*/(short)0, c, /*reuse_a=*/false, /*reuse_b=*/false);
}

// A-fragment (16-bit A 16x32): for lane L, halves 0..15 hold
// K = 2*b8..2*b8+15 (contiguous), b8 = 8*(L>=16),
// row M = L%16.  `row` points at A[M][0], contiguous over K.
__device__ __forceinline__ v16h a_frag(const half_t* row, int k0, int b8) {
  v8h lo = *(const v8h*)(row + k0 + 2 * b8);
  v8h hi = *(const v8h*)(row + k0 + 2 * b8 + 8);
  v16h a;
#pragma unroll
  for (int i = 0; i < 8; ++i) { a[i] = lo[i]; a[8 + i] = hi[i]; }
  return a;
}

// s_wait_asynccnt 0 (ASYNCcnt drains the async-to-LDS queue)
__device__ __forceinline__ void wait_async0() {
#if __has_builtin(__builtin_amdgcn_s_wait_asynccnt)
  __builtin_amdgcn_s_wait_asynccnt(0);
#else
  asm volatile("s_wait_asynccnt 0x0" ::: "memory");
#endif
}

// Low 32 bits of a generic pointer to __shared__ = LDS byte offset
// (AMDGPU generic shared pointers are {shared_aperture_hi32, lds_offset}).
__device__ __forceinline__ unsigned lds_offset_of(const void* p) {
  return (unsigned)(unsigned long long)p;
}

// Gather the 16x576 im2col patch matrix xu[:, :, l] into LDS as f16.
// i = c*9 + kh*3 + kw (torch Unfold order), l = y*32 + x, pad=1.
__device__ __forceinline__ void fill_patch_lds(const float* __restrict__ x,
                                               half_t* Alds, int l) {
  const int y  = l >> 5;
  const int xx = l & 31;
#pragma unroll
  for (int it = 0; it < 36; ++it) {           // 16*576 / 256 = 36
    int f = threadIdx.x + it * 256;
    int b = f / CKKv;
    int i = f - b * CKKv;
    int c = i / 9;
    int r = i - c * 9;
    int kh = r / 3, kw = r - kh * 3;
    int iy = y + kh - 1, ix = xx + kw - 1;
    float v = 0.f;
    if ((unsigned)iy < 32u && (unsigned)ix < 32u)
      v = x[(((b * Cv) + c) * Hv + iy) * Wv + ix];
    Alds[f] = (half_t)v;
  }
}

// ---------------------------------------------------------------------------
// Prologue: f32 -> f16 copy of the base kernel (enables contiguous B loads)
// ---------------------------------------------------------------------------
__global__ void __launch_bounds__(256)
k_convert(const float* __restrict__ src, half_t* __restrict__ dst, int n) {
  int i = blockIdx.x * 256 + threadIdx.x;
  if (i < n) dst[i] = (half_t)src[i];
}

// ---------------------------------------------------------------------------
// MLP table: Hh[r][j] = silu(W1[j]·[gx/32, gy/32, seed] + b1[j]),
// r = gx*32 + gy (meshgrid indexing='ij' over (W,H) per the reference).
// ---------------------------------------------------------------------------
__global__ void __launch_bounds__(256)
k_mlp(const float* __restrict__ seed, const float* __restrict__ W1,
      const float* __restrict__ b1, half_t* __restrict__ Hh) {
  const int r = blockIdx.x * 256 + threadIdx.x;
  if (r >= Lv) return;
  const float in0 = (float)(r >> 5) * 0.03125f;   // gx / ow
  const float in1 = (float)(r & 31) * 0.03125f;   // gy / oh
  float sd[16];
#pragma unroll
  for (int d = 0; d < 16; ++d) sd[d] = seed[d];
#pragma unroll 4
  for (int j = 0; j < H2v; ++j) {
    const float* w = W1 + j * 18;
    float z = b1[j] + w[0] * in0 + w[1] * in1;
#pragma unroll
    for (int d = 0; d < 16; ++d) z += w[2 + d] * sd[d];
    float h = z / (1.f + expf(-z));               // silu
    Hh[r * H2v + j] = (half_t)h;
  }
}

// ---------------------------------------------------------------------------
// Stage A: G[b,l,k] = sum_i xu[b,i,l] * W2[(l%128)*576 + i, k]
// One block per location l.  The 72 KB contiguous W2 slice for this block is
// staged into LDS in two 36 KB chunks with GLOBAL_LOAD_ASYNC_TO_LDS_B128
// (ASYNCcnt); waves 0,1 run the WMMAs (one 16-wide k tile each), all 8 waves
// drive the async copy and barriers.
// ---------------------------------------------------------------------------
#define GCHUNK 288                                // i-rows per chunk (2 chunks)

__global__ void __launch_bounds__(256)
k_G(const float* __restrict__ x, const float* __restrict__ W2,
    half_t* __restrict__ G) {
  __shared__ __align__(16) half_t Alds[Bv * CKKv];       // 18 KB
  __shared__ __align__(16) float  W2lds[GCHUNK * H2v];   // 36 KB
  const int l = blockIdx.x;
  fill_patch_lds(x, Alds, l);

  const int wave = threadIdx.x >> 5;
  const int lane = threadIdx.x & 31;
  const int hiL  = lane >> 4;
  const int m16  = lane & 15;
  const int b8   = hiL * 8;
  const int kk0  = wave * 16;
  const half_t* Arow = Alds + m16 * CKKv;
  const float*  gsrc = W2 + (size_t)(l & 127) * CKKv * H2v;  // contiguous slice
  const unsigned ldsb = lds_offset_of(W2lds);

  v8f acc = {};
  for (int chunk = 0; chunk < 2; ++chunk) {
    // --- async bulk copy: GCHUNK*32 floats = 2304 float4 / 256 threads ---
#pragma unroll
    for (int j = 0; j < 9; ++j) {
      const unsigned idx = (unsigned)(j * 256 + threadIdx.x);   // float4 index
      const float* gp = gsrc + (size_t)(chunk * GCHUNK * H2v) + idx * 4;
#if USE_ASYNC_LDS
      unsigned ldst = ldsb + idx * 16u;
      asm volatile("global_load_async_to_lds_b128 %0, %1, off"
                   :: "v"(ldst), "v"(gp) : "memory");
#else
      *(float4*)((char*)W2lds + idx * 16u) = *(const float4*)gp;
#endif
    }
#if USE_ASYNC_LDS
    wait_async0();
#endif
    __syncthreads();                              // chunk visible to all waves

    if (wave < 2) {                               // wave-uniform: EXEC all-1s
#pragma unroll 3
      for (int s = 0; s < GCHUNK / 32; ++s) {     // 9 WMMA steps per chunk
        const int i0 = chunk * GCHUNK + s * 32;
        v16h a = a_frag(Arow, i0, b8);
        v16h bf;                                  // B[kr][n] from LDS tile
        const float* bp = W2lds + (s * 32 + 16 * hiL) * H2v + (kk0 + m16);
#pragma unroll
        for (int h = 0; h < 16; ++h) bf[h] = (half_t)bp[h * H2v];
        acc = wmma16x16x32(a, bf, acc);
      }
    }
    __syncthreads();                              // done reading before rewrite
  }

  if (wave < 2) {
    const int k = kk0 + m16;
#pragma unroll
    for (int v = 0; v < 8; ++v) {                 // C: M = v + b8 (= batch)
      int bb = v + b8;
      G[((bb * Lv) + l) * H2v + k] = (half_t)acc[v];
    }
  }
}

// ---------------------------------------------------------------------------
// Stage B: out[b,o,l] = sum_i xu[b,i,l]*kernel[o,i]            (18 WMMA)
//                      + sum_k Hh[8o + l/128, k]*G[b,l,k]       (1 WMMA)
//                      + sum_i xu[b,i,l]*b2[(l%128)*576+i]      (VALU)
//                      + bias[o]
// One block per location l; 8 waves, each owns one 16-output tile.
// ---------------------------------------------------------------------------
__global__ void __launch_bounds__(256)
k_out(const float* __restrict__ x, const half_t* __restrict__ kf16,
      const float* __restrict__ bias, const float* __restrict__ b2,
      const half_t* __restrict__ Hh, const half_t* __restrict__ G,
      float* __restrict__ out) {
  __shared__ __align__(16) half_t Alds[Bv * CKKv];       // 18 KB
  __shared__ float  sE[256];
  __shared__ float  Eb[16];
  const int l = blockIdx.x;
  fill_patch_lds(x, Alds, l);
  __syncthreads();

  // E[b] = xu[b,:,l] · b2[(l%128)*576 + :]
  {
    const int b   = threadIdx.x & 15;
    const int seg = threadIdx.x >> 4;
    const float* b2row = b2 + (l & 127) * CKKv;
    float p = 0.f;
    const int i0 = seg * 36;
#pragma unroll 4
    for (int i = i0; i < i0 + 36; ++i)
      p += (float)Alds[b * CKKv + i] * b2row[i];
    sE[threadIdx.x] = p;
  }
  __syncthreads();
  if (threadIdx.x < 16) {
    float e = 0.f;
#pragma unroll
    for (int seg = 0; seg < 16; ++seg) e += sE[seg * 16 + threadIdx.x];
    Eb[threadIdx.x] = e;
  }
  __syncthreads();

  const int wave = threadIdx.x >> 5;
  const int lane = threadIdx.x & 31;
  const int hiL  = lane >> 4;
  const int m16  = lane & 15;
  const int b8   = hiL * 8;
  const int o0   = wave * 16;
  const half_t* Arow = Alds + m16 * CKKv;
  const half_t* Brow = kf16 + (o0 + m16) * CKKv + 16 * hiL;  // B[kr][n]=kernel[o0+n][i0+kr]

  v8f acc = {};
#pragma unroll 2
  for (int i0 = 0; i0 < CKKv; i0 += 32) {         // main GEMM: 18 WMMA, K = 576
    v16h a  = a_frag(Arow, i0, b8);
    v16h bf = *(const v16h*)(Brow + i0);          // 32B-aligned contiguous K16
    acc = wmma16x16x32(a, bf, acc);
  }
  {                                               // position-MLP correction, K = 32
    const int lg = l >> 7;
    v16h a2 = a_frag(G + ((size_t)m16 * Lv + l) * H2v, 0, b8);
    v16h bh = *(const v16h*)(Hh + ((o0 + m16) * 8 + lg) * H2v + 16 * hiL);
    acc = wmma16x16x32(a2, bh, acc);
  }

  const int o  = o0 + m16;
  const float bo = bias[o];
#pragma unroll
  for (int v = 0; v < 8; ++v) {                   // C: M = v + b8 (= batch)
    int bb = v + b8;
    out[((bb * Ov) + o) * Lv + l] = acc[v] + bo + Eb[bb];
  }
}

// ---------------------------------------------------------------------------
// Host launcher
// ---------------------------------------------------------------------------
extern "C" void kernel_launch(void* const* d_in, const int* in_sizes, int n_in,
                              void* d_out, int out_size, void* d_ws, size_t ws_size,
                              hipStream_t stream) {
  (void)in_sizes; (void)n_in; (void)out_size; (void)ws_size;
  const float* x    = (const float*)d_in[0];
  const float* kern = (const float*)d_in[1];
  const float* bias = (const float*)d_in[2];
  const float* seed = (const float*)d_in[3];
  const float* W1   = (const float*)d_in[4];
  const float* b1   = (const float*)d_in[5];
  const float* W2   = (const float*)d_in[6];
  const float* b2   = (const float*)d_in[7];
  float* out = (float*)d_out;

  // Workspace layout (all fully rewritten every call; ~1.2 MB total):
  //   [0, 147456)        kernel as f16   (128*576)
  //   [147456, 212992)   Hh MLP table f16 (1024*32)
  //   [212992, 1261568)  G f16            (16*1024*32)
  char* ws = (char*)d_ws;
  half_t* kf16 = (half_t*)(ws);
  half_t* Hh   = (half_t*)(ws + 147456);
  half_t* G    = (half_t*)(ws + 212992);

  k_convert<<<dim3((Ov * CKKv + 255) / 256), dim3(256), 0, stream>>>(kern, kf16, Ov * CKKv);
  k_mlp    <<<dim3(Lv / 256),                dim3(256), 0, stream>>>(seed, W1, b1, Hh);
  k_G      <<<dim3(Lv),                      dim3(256), 0, stream>>>(x, W2, G);
  k_out    <<<dim3(Lv),                      dim3(256), 0, stream>>>(x, kf16, bias, b2, Hh, G, out);
}
